// NN_CYK_Model_26671746908679
// MI455X (gfx1250) — matrix operand order repaired and checked by reading, not compiled, not run on hardware
//
#include <hip/hip_runtime.h>
#include <hip/hip_bf16.h>
#include <stdint.h>

// ---------- problem constants ----------
#define CB   64      // batch
#define CL   32      // sentence length
#define CNT  64      // nonterminals
#define CT   64      // preterminals
#define CS   128     // NT+T symbols
#define CV   50000   // vocab
#define CEMB 512
#define CSD  512
#define CEPS 1e-9f

typedef float v8f  __attribute__((ext_vector_type(8)));
typedef __bf16 v16bf __attribute__((ext_vector_type(16)));

union AFrag { v16bf v; __bf16 e[16]; uint4 u[2]; };

__device__ __forceinline__ v8f wmma_bf16(v16bf a, v16bf b, v8f c) {
  // D = A(16x32 bf16) * B(32x16 bf16) + C(16x16 f32)
  return __builtin_amdgcn_wmma_f32_16x16x32_bf16(false, a, false, b, (short)0, c,
                                                 false, false);
}

// ================= packing kernels =================

// G (NT,S,S) f32 -> G2packed bf16 in B-fragment layout:
// G2p[c][at][lane][j], k = c*32 + 16*(lane>>4) + j, col a = at*16 + (lane&15),
// G2[k,a] = G[a, k>>7, k&127]
__global__ void pack_G2(const float* __restrict__ G, __bf16* __restrict__ G2p) {
  int c    = blockIdx.x;            // 0..511 k-chunks
  int at   = blockIdx.y;            // 0..3 a-tiles
  int lane = threadIdx.x;
  int a    = at * 16 + (lane & 15);
  int kb   = c * 32 + ((lane >> 4) << 4);
  __bf16* dst = G2p + (((size_t)(c * 4 + at) * 32 + lane) * 16);
#pragma unroll
  for (int j = 0; j < 16; ++j) {
    int k = kb + j;
    int s = k >> 7, t = k & 127;
    dst[j] = (__bf16)G[((size_t)a * CS + s) * CS + t];
  }
}

// W (K x 512) f32 -> fragment layout Wpk[kc][nt][lane][16]
__global__ void pack_W(const float* __restrict__ W, __bf16* __restrict__ Wpk) {
  int kc   = blockIdx.x;            // k-chunk
  int nt   = blockIdx.y;            // 0..31 n-tiles
  int lane = threadIdx.x;
  int col  = nt * 16 + (lane & 15);
  int kb   = kc * 32 + ((lane >> 4) << 3);
  __bf16* dst = Wpk + (((size_t)(kc * 32 + nt) * 32 + lane) * 16);
#pragma unroll
  for (int j = 0; j < 16; ++j)
    dst[j] = (__bf16)W[(size_t)(kb + (j & 7) + ((j >> 3) << 4)) * CSD + col];
}

// embB[m][e] = bf16(word_embeddings[word[m]][e]),  m = b*L + l  (2048 rows)
__global__ void gather_emb(const int* __restrict__ word,
                           const float* __restrict__ emb,
                           __bf16* __restrict__ embB) {
  int m = blockIdx.x;
  int w = word[m];
  for (int e = threadIdx.x; e < CEMB; e += blockDim.x)
    embB[(size_t)m * CEMB + e] = (__bf16)emb[(size_t)w * CEMB + e];
}

// ================= feat0 = relu(emb @ Wp + bp) -> chartF depth 0 =============

__global__ __launch_bounds__(32)
void feat0_gemm(const __bf16* __restrict__ embB, const __bf16* __restrict__ WpP,
                const float* __restrict__ bp, __bf16* __restrict__ chartF) {
  int m0   = blockIdx.x * 16;       // 128 row tiles
  int nt   = blockIdx.y;            // 32 col tiles
  int lane = threadIdx.x;
  int row  = lane & 15, half = lane >> 4;
  const __bf16* arow = embB + (size_t)(m0 + row) * CEMB;
  v8f c = {};
  for (int kc = 0; kc < 16; ++kc) {
    AFrag af;
    af.u[0] = *(const uint4*)(arow + kc * 32 + 8 * half);
    af.u[1] = *(const uint4*)(arow + kc * 32 + 16 + 8 * half);
    v16bf bfr = *(const v16bf*)(WpP + ((size_t)(kc * 32 + nt) * 32 + lane) * 16);
    c = wmma_bf16(af.v, bfr, c);
  }
  int col = nt * 16 + (lane & 15);
  float bpv = bp[col];
#pragma unroll
  for (int r = 0; r < 8; ++r) {
    int gm = m0 + r + 8 * half;     // gm = b*32 + l; chart index (b,l,depth0)
    float v = c[r] + bpv;
    v = v > 0.f ? v : 0.f;
    chartF[((size_t)gm * CL) * CSD + col] = (__bf16)v;
  }
}

// ================= diag_p -> chartP depth 0 =================

__global__ void diag_p_kernel(const int* __restrict__ word,
                              const float* __restrict__ preterm,
                              __bf16* __restrict__ chartP) {
  int cell = blockIdx.x;            // b*L + l
  int s    = threadIdx.x;           // 0..127
  int w    = word[cell];
  float v  = (s >= CNT) ? preterm[(size_t)(s - CNT) * CV + w] : 0.f;
  __shared__ float red[CS];
  red[s] = v;
  __syncthreads();
  for (int off = 64; off > 0; off >>= 1) {
    if (s < off) red[s] += red[s + off];
    __syncthreads();
  }
  float sum = red[0] + CEPS;
  chartP[((size_t)cell * CL) * CS + s] = (__bf16)(v / sum);
}

__global__ void zeroF(float* p, int count) {
  int i = blockIdx.x * blockDim.x + threadIdx.x;
  if (i < count) p[i] = 0.f;
}

// ================= CYK score kernel =================
// s_d[m,a] = sum_{s,t} lp[m,s] rp[m,t] G[a,s,t]  via  Y(m,k)=lp*rp , Y @ G2
// grid: (M/16, ln-1), 1 wave per block. lp/rp rows staged into LDS with
// GLOBAL_LOAD_ASYNC_TO_LDS_B128 (ASYNCcnt). A fragment built with two
// ds_read_b128 + packed bf16 multiply against a broadcast lp value.
// Accumulates scores over d with atomics; ds-reduces per-row masses.
__global__ __launch_bounds__(32)
void cyk_scores(const __bf16* __restrict__ chartP, const __bf16* __restrict__ G2p,
                float* __restrict__ scores, float* __restrict__ masses,
                int n, int ln, int M) {
  int bx   = blockIdx.x;
  int d    = blockIdx.y + 1;
  int lane = threadIdx.x;
  __shared__ __bf16 sLP[16 * CS];
  __shared__ __bf16 sRP[16 * CS];
  __shared__ float  sMass[16];

  // stage 16 lp rows + 16 rp rows (bf16, 256B each) into LDS asynchronously
  uint32_t ldsL = (uint32_t)(uintptr_t)sLP;
  uint32_t ldsR = (uint32_t)(uintptr_t)sRP;
  for (int idx = lane; idx < 256; idx += 32) {
    int r = idx >> 4, q = idx & 15;
    int gm = bx * 16 + r;
    int b = gm / n, i = gm - b * n;
    const uint4* lsrc =
        (const uint4*)(chartP + ((size_t)((b * CL + i) * CL + (d - 1))) * CS) + q;
    const uint4* rsrc =
        (const uint4*)(chartP + ((size_t)((b * CL + i + d) * CL + (ln - d - 1))) * CS) + q;
    uint32_t dl = ldsL + idx * 16;
    uint32_t dr = ldsR + idx * 16;
    asm volatile("global_load_async_to_lds_b128 %0, %1, off"
                 :: "v"(dl), "v"(lsrc) : "memory");
    asm volatile("global_load_async_to_lds_b128 %0, %1, off"
                 :: "v"(dr), "v"(rsrc) : "memory");
  }
  asm volatile("s_wait_asynccnt 0x0" ::: "memory");
  __syncthreads();

  int row = lane & 15, half = lane >> 4;
  const char* rbase = (const char*)sRP + row * 256 + 32 * half; // bytes
  const unsigned short* lrow = (const unsigned short*)sLP + row * CS;

  v8f c0 = {}, c1 = {}, c2 = {}, c3 = {};
  const __bf16* gb = G2p;
  for (int s = 0; s < CS; ++s) {            // lp symbol index
    uint32_t lr  = lrow[s];
    uint32_t lpp = lr | (lr << 16);         // broadcast bf16 pair
    AFrag lvec;
    lvec.u[0] = make_uint4(lpp, lpp, lpp, lpp);
    lvec.u[1] = lvec.u[0];
#pragma unroll
    for (int tt = 0; tt < 4; ++tt) {        // rp 32-wide sub-chunks
      AFrag rvec;
      rvec.u[0] = *(const uint4*)(rbase + tt * 64);
      rvec.u[1] = *(const uint4*)(rbase + tt * 64 + 16);
      AFrag af;
      af.v = lvec.v * rvec.v;               // packed bf16 multiply
      v16bf b0 = *(const v16bf*)(gb + ((size_t)0 * 32 + lane) * 16);
      v16bf b1 = *(const v16bf*)(gb + ((size_t)1 * 32 + lane) * 16);
      v16bf b2 = *(const v16bf*)(gb + ((size_t)2 * 32 + lane) * 16);
      v16bf b3 = *(const v16bf*)(gb + ((size_t)3 * 32 + lane) * 16);
      gb += 4 * 32 * 16;
      c0 = wmma_bf16(af.v, b0, c0);
      c1 = wmma_bf16(af.v, b1, c1);
      c2 = wmma_bf16(af.v, b2, c2);
      c3 = wmma_bf16(af.v, b3, c3);
    }
  }

  // accumulate scores over d (atomic f32), reduce masses for this d
#pragma unroll
  for (int r = 0; r < 8; ++r) {
    int gm = bx * 16 + r + 8 * half;
    int a0 = lane & 15;
    unsafeAtomicAdd(&scores[(size_t)gm * CNT + a0],      c0[r]);
    unsafeAtomicAdd(&scores[(size_t)gm * CNT + 16 + a0], c1[r]);
    unsafeAtomicAdd(&scores[(size_t)gm * CNT + 32 + a0], c2[r]);
    unsafeAtomicAdd(&scores[(size_t)gm * CNT + 48 + a0], c3[r]);
  }
  if (lane < 16) sMass[lane] = 0.f;
  __syncthreads();
#pragma unroll
  for (int r = 0; r < 8; ++r)
    atomicAdd(&sMass[r + 8 * half], c0[r] + c1[r] + c2[r] + c3[r]);
  __syncthreads();
  if (lane < 16) masses[(size_t)(d - 1) * M + bx * 16 + lane] = sMass[lane];
}

// ================= normalize p, compute weight denominators =================

__global__ void cyk_norm(__bf16* __restrict__ chartP, const float* __restrict__ scores,
                         const float* __restrict__ masses, float* __restrict__ wden,
                         int n, int ln, int M) {
  int m = blockIdx.x;
  int a = threadIdx.x;              // 0..127
  __shared__ float red[CS];
  float v = (a < CNT) ? scores[(size_t)m * CNT + a] : 0.f;
  red[a] = v;
  __syncthreads();
  for (int off = 64; off > 0; off >>= 1) {
    if (a < off) red[a] += red[a + off];
    __syncthreads();
  }
  float s = red[0] + CEPS;
  int b = m / n, i = m - b * n;
  chartP[((size_t)((b * CL + i) * CL + (ln - 1))) * CS + a] =
      (a < CNT) ? (__bf16)(v / s) : (__bf16)0.f;
  if (a == 0) {
    float sd = 0.f;
    for (int d = 1; d < ln; ++d) sd += masses[(size_t)(d - 1) * M + m];
    wden[m] = sd + CEPS;
  }
}

// ================= feature kernel: f = sum_d w_d * relu([lf,rf]@Wm + bm) =====

__global__ __launch_bounds__(32)
void cyk_feats(__bf16* __restrict__ chartF, const __bf16* __restrict__ WmP,
               const float* __restrict__ bm, const float* __restrict__ masses,
               const float* __restrict__ wden, int n, int ln, int M) {
  int bx   = blockIdx.x;            // m tile
  int hy   = blockIdx.y;            // 0..31 output h tile
  int lane = threadIdx.x;
  int row  = lane & 15, half = lane >> 4;
  int gmA  = bx * 16 + row;
  int bA   = gmA / n, iA = gmA - bA * n;
  int col  = hy * 16 + (lane & 15);
  float bmv = bm[col];
  float wd[8];
#pragma unroll
  for (int r = 0; r < 8; ++r) wd[r] = wden[bx * 16 + r + 8 * half];

  v8f facc = {};
  for (int d = 1; d < ln; ++d) {
    const __bf16* lf =
        chartF + ((size_t)((bA * CL + iA) * CL + (d - 1))) * CSD;
    const __bf16* rf =
        chartF + ((size_t)((bA * CL + iA + d) * CL + (ln - d - 1))) * CSD;
    __builtin_prefetch(rf, 0, 0);   // global_prefetch_b8
    v8f c = {};
#pragma unroll 4
    for (int kc = 0; kc < 32; ++kc) {   // K = 1024 (concat lf|rf)
      const __bf16* srow = (kc < 16) ? lf : rf;
      int k0 = (kc & 15) * 32;
      AFrag af;
      af.u[0] = *(const uint4*)(srow + k0 + 8 * half);
      af.u[1] = *(const uint4*)(srow + k0 + 16 + 8 * half);
      v16bf bfr = *(const v16bf*)(WmP + ((size_t)(kc * 32 + hy) * 32 + lane) * 16);
      c = wmma_bf16(af.v, bfr, c);
    }
#pragma unroll
    for (int r = 0; r < 8; ++r) {
      int gm = bx * 16 + r + 8 * half;
      float w = masses[(size_t)(d - 1) * M + gm] / wd[r];
      float v = c[r] + bmv;
      v = v > 0.f ? v : 0.f;
      facc[r] += w * v;
    }
  }
#pragma unroll
  for (int r = 0; r < 8; ++r) {
    int gm = bx * 16 + r + 8 * half;
    int b = gm / n, i = gm - b * n;
    chartF[((size_t)((b * CL + i) * CL + (ln - 1))) * CSD + col] = (__bf16)facc[r];
  }
}

// ================= root =================

__global__ void root_kernel(const __bf16* __restrict__ chartP,
                            const __bf16* __restrict__ chartF,
                            const float* __restrict__ starts,
                            float* __restrict__ out) {
  int b = blockIdx.x;
  int t = threadIdx.x;              // 128
  __shared__ float red[CS];
  float v = (float)chartP[((size_t)((b * CL + 0) * CL + (CL - 1))) * CS + t] * starts[t];
  red[t] = v;
  __syncthreads();
  for (int off = 64; off > 0; off >>= 1) {
    if (t < off) red[t] += red[t + off];
    __syncthreads();
  }
  float rs = red[0];
  for (int h = t; h < CSD; h += 128)
    out[(size_t)b * CSD + h] =
        (float)chartF[((size_t)((b * CL) * CL + (CL - 1))) * CSD + h] * rs;
}

// ================= host launcher =================

extern "C" void kernel_launch(void* const* d_in, const int* in_sizes, int n_in,
                              void* d_out, int out_size, void* d_ws, size_t ws_size,
                              hipStream_t stream) {
  const int*   word    = (const int*)d_in[0];
  const float* wembed  = (const float*)d_in[1];
  const float* preterm = (const float*)d_in[2];
  const float* G       = (const float*)d_in[3];
  const float* starts  = (const float*)d_in[4];
  const float* Wp      = (const float*)d_in[5];
  const float* bp      = (const float*)d_in[6];
  const float* Wm      = (const float*)d_in[7];
  const float* bm      = (const float*)d_in[8];
  float* out = (float*)d_out;

  // ---- workspace carve (all regions 256B aligned) ----
  char* ws = (char*)d_ws;
  size_t off = 0;
  auto carve = [&](size_t bytes) {
    char* p = ws + off;
    off += (bytes + 255) & ~(size_t)255;
    return p;
  };
  __bf16* chartP = (__bf16*)carve((size_t)CB * CL * CL * CS * 2);    // 16.8 MB
  __bf16* chartF = (__bf16*)carve((size_t)CB * CL * CL * CSD * 2);   // 67 MB
  __bf16* G2p    = (__bf16*)carve((size_t)512 * 4 * 32 * 16 * 2);    // 2 MB
  __bf16* WmP    = (__bf16*)carve((size_t)32 * 32 * 32 * 16 * 2);    // 1 MB
  __bf16* WpP    = (__bf16*)carve((size_t)16 * 32 * 32 * 16 * 2);    // 0.5 MB
  __bf16* embB   = (__bf16*)carve((size_t)CB * CL * CEMB * 2);       // 2 MB
  float*  scores = (float*)carve((size_t)CB * 31 * CNT * 4);
  float*  masses = (float*)carve((size_t)31 * CB * 31 * 4);
  float*  wden   = (float*)carve((size_t)CB * 31 * 4);

  // ---- packing / prologue ----
  pack_G2<<<dim3(512, 4), 32, 0, stream>>>(G, G2p);
  pack_W<<<dim3(32, 32), 32, 0, stream>>>(Wm, WmP);
  pack_W<<<dim3(16, 32), 32, 0, stream>>>(Wp, WpP);
  gather_emb<<<CB * CL, 256, 0, stream>>>(word, wembed, embB);
  feat0_gemm<<<dim3((CB * CL) / 16, CSD / 16), 32, 0, stream>>>(embB, WpP, bp, chartF);
  diag_p_kernel<<<CB * CL, CS, 0, stream>>>(word, preterm, chartP);

  // ---- CYK chart loop ----
  for (int ln = 2; ln <= CL; ++ln) {
    int n = CL - ln + 1;
    int M = CB * n;                 // rows (cells) in this diagonal
    int mt = M / 16;
    zeroF<<<(M * CNT + 255) / 256, 256, 0, stream>>>(scores, M * CNT);
    cyk_scores<<<dim3(mt, ln - 1), 32, 0, stream>>>(chartP, G2p, scores, masses,
                                                    n, ln, M);
    cyk_norm<<<M, CS, 0, stream>>>(chartP, scores, masses, wden, n, ln, M);
    cyk_feats<<<dim3(mt, CSD / 16), 32, 0, stream>>>(chartF, WmP, bm, masses,
                                                     wden, n, ln, M);
  }

  // ---- root ----
  root_kernel<<<CB, CS, 0, stream>>>(chartP, chartF, starts, out);
}
